// SingleLayerTransformer_23545010717466
// MI455X (gfx1250) — matrix operand order repaired and checked by reading, not compiled, not run on hardware
//
#include <hip/hip_runtime.h>
#include <math.h>

// ---------------------------------------------------------------------------
// Types for CDNA5 WMMA (wave32). v_wmma_f32_16x16x32_bf16: D = A(16x32) x B(32x16) + C
// ---------------------------------------------------------------------------
typedef __attribute__((ext_vector_type(16))) __bf16 v16bf;
typedef __attribute__((ext_vector_type(8)))  float  v8f;
typedef __attribute__((ext_vector_type(4)))  unsigned int u32x4;
typedef __attribute__((ext_vector_type(8)))  int i32x8;
typedef __attribute__((ext_vector_type(4)))  int i32x4;

struct __align__(16) U4 { unsigned int x, y, z, w; };
union Frag { U4 u[2]; v16bf v; };

// Round-to-nearest-even f32 -> bf16 (stored as raw u16)
static __device__ __forceinline__ unsigned short f2b(float f) {
    unsigned int u = __float_as_uint(f);
    u += 0x7FFFu + ((u >> 16) & 1u);
    return (unsigned short)(u >> 16);
}

// ---------------------------------------------------------------------------
// Model dims
// ---------------------------------------------------------------------------
#define SEQ   4096
#define DM    512
#define D3    1536
#define DF    2048
#define NH    8
#define HD    64
#define NC    4
#define CH    1024
#define VOC   32000
#define BK    64          // staged K per TDM tile (two WMMA k-chunks)

// ---------------------------------------------------------------------------
// Tensor Data Mover: DMA a 2-D bf16 tile (tile_rows x tile_k elems) from
// global (row stride = src_stride elems) into LDS at byte offset lds_off,
// rows packed contiguously (tile_k elems each). D# per CDNA5 ISA.
// ---------------------------------------------------------------------------
static __device__ __forceinline__ void tdm_load_2d(const unsigned short* gptr,
                                                   unsigned int lds_off,
                                                   unsigned int tile_k,
                                                   unsigned int tile_rows,
                                                   unsigned int src_stride) {
    const unsigned long long ga = (unsigned long long)(uintptr_t)gptr;
    u32x4 g0;
    g0[0] = 1u;                                            // count=1, user descriptor
    g0[1] = lds_off;                                       // lds_addr [63:32]
    g0[2] = (unsigned int)(ga & 0xFFFFFFFFu);              // global_addr[31:0]
    g0[3] = (unsigned int)((ga >> 32) & 0x01FFFFFFu)       // global_addr[56:32]
          | (2u << 30);                                    // type=2 ("image")
    const unsigned int dim0 = src_stride;                  // tensor_dim0 (OOB bound)
    const unsigned int dim1 = 0x40000000u;                 // tensor_dim1: unbounded rows
    const unsigned long long s0 = (unsigned long long)src_stride;  // dim0 stride
    i32x8 g1;
    g1[0] = (int)(1u << 16);                               // data_size=2B; mask/pad/iter=0
    g1[1] = (int)((dim0 & 0xFFFFu) << 16);                 // [15:0] atomic_barrier_addr=0
    g1[2] = (int)((dim0 >> 16) | ((dim1 & 0xFFFFu) << 16));
    g1[3] = (int)((dim1 >> 16) | (tile_k << 16));          // tile_dim0
    g1[4] = (int)tile_rows;                                // tile_dim1 (tile_dim2=0)
    g1[5] = (int)(unsigned int)(s0 & 0xFFFFFFFFu);         // tensor_dim0_stride[31:0]
    g1[6] = (int)(unsigned int)((s0 >> 32) & 0xFFFFu);     // stride[47:32]; dim1_stride=0
    g1[7] = 0;
    const i32x4 z4 = {0, 0, 0, 0};
#if defined(__clang_major__) && __clang_major__ >= 23
    const i32x8 z8 = {0, 0, 0, 0, 0, 0, 0, 0};
    __builtin_amdgcn_tensor_load_to_lds(g0, g1, z4, z4, z8, 0);
#else
    __builtin_amdgcn_tensor_load_to_lds(g0, g1, z4, z4, 0);
#endif
}

// ---------------------------------------------------------------------------
// WMMA GEMM, double-buffered TDM->LDS pipeline.
//   C[m][n] = scale*sum_k A[m][k]*B[n][k] + bias[n]
// A: M x K bf16 row-major (lda), B: N x K bf16 row-major (ldb).
// grid = (N/64, M/128), block = 256 = 8 waves (4 M-waves x 2 N-waves),
// wave tile 32x32 (2x2 WMMA tiles), staged K = 64 (8 WMMAs per barrier pair).
// Wave 0 DMAs tile t+1 while all 8 waves compute tile t; "tensorcnt<=2"
// guarantees tile t's two descriptors completed (TDM is in-order per wave).
// ---------------------------------------------------------------------------
__global__ void __launch_bounds__(256)
k_gemm_bf16(const unsigned short* __restrict__ A, int lda,
            const unsigned short* __restrict__ B, int ldb,
            const float* __restrict__ bias,
            float* __restrict__ Cf,
            unsigned short* __restrict__ Cb, int ldc,
            int K, float scale, int relu) {
    __shared__ __align__(16) unsigned short Alds[2][128 * BK];   // 32 KB
    __shared__ __align__(16) unsigned short Blds[2][64 * BK];    // 16 KB

    const int wid  = threadIdx.x >> 5;
    const int lane = threadIdx.x & 31;
    const int l    = lane & 15;
    const int hi   = lane >> 4;          // half-wave select
    const int wm   = wid & 3;            // 4 wave-rows   (32 M each)
    const int wn   = wid >> 2;           // 2 wave-cols   (32 N each)

    const int blkM = blockIdx.y * 128;
    const int blkN = blockIdx.x * 64;

    // Per-lane LDS fragment element offsets (rows packed as BK elems = 128B).
    // A frag: lanes<16 hold K {0..15}; lanes>=16 hold K {16..31}.
    const int aBase0 = (wm * 32 + l)      * BK + (hi ? 16 : 0);
    const int aBase1 = (wm * 32 + 16 + l) * BK + (hi ? 16 : 0);
    // B frag: lanes<16 hold K {0..15}; lanes>=16 hold K {16..31}.
    const int bBase0 = (wn * 32 + l)      * BK + (hi ? 16 : 0);
    const int bBase1 = (wn * 32 + 16 + l) * BK + (hi ? 16 : 0);

    const unsigned int a_off[2] = { (unsigned int)(uintptr_t)&Alds[0][0],
                                    (unsigned int)(uintptr_t)&Alds[1][0] };
    const unsigned int b_off[2] = { (unsigned int)(uintptr_t)&Blds[0][0],
                                    (unsigned int)(uintptr_t)&Blds[1][0] };

    v8f acc00 = {0.f,0.f,0.f,0.f,0.f,0.f,0.f,0.f};
    v8f acc01 = acc00, acc10 = acc00, acc11 = acc00;

    const int T = K / BK;
    if (wid == 0) {   // prologue: stage tile 0 into buffer 0
        tdm_load_2d(A + (size_t)blkM * lda, a_off[0], BK, 128u, (unsigned)lda);
        tdm_load_2d(B + (size_t)blkN * ldb, b_off[0], BK,  64u, (unsigned)ldb);
    }

    for (int t = 0; t < T; ++t) {
        const int buf = t & 1;
        if (wid == 0) {
            if (t + 1 < T) {
                // Next buffer was consumed at t-1; end-of-iteration barrier of
                // t-1 makes it safe to overwrite. Issue before waiting so the
                // DMA overlaps this tile's compute.
                const int nb = (t + 1) & 1;
                tdm_load_2d(A + (size_t)blkM * lda + (size_t)(t + 1) * BK,
                            a_off[nb], BK, 128u, (unsigned)lda);
                tdm_load_2d(B + (size_t)blkN * ldb + (size_t)(t + 1) * BK,
                            b_off[nb], BK, 64u, (unsigned)ldb);
                __builtin_amdgcn_s_wait_tensorcnt((short)2);  // tile t landed
            } else {
                __builtin_amdgcn_s_wait_tensorcnt((short)0);
            }
        }
        __syncthreads();                        // publish LDS tile t

        const unsigned short* Ab = &Alds[buf][0];
        const unsigned short* Bb = &Blds[buf][0];
        #pragma unroll
        for (int ku = 0; ku < BK; ku += 32) {
            Frag a0, a1, b0, b1;
            a0.u[0] = *(const U4*)(Ab + aBase0 + ku);
            a0.u[1] = *(const U4*)(Ab + aBase0 + ku + 8);
            a1.u[0] = *(const U4*)(Ab + aBase1 + ku);
            a1.u[1] = *(const U4*)(Ab + aBase1 + ku + 8);
            b0.u[0] = *(const U4*)(Bb + bBase0 + ku);
            b0.u[1] = *(const U4*)(Bb + bBase0 + ku + 8);
            b1.u[0] = *(const U4*)(Bb + bBase1 + ku);
            b1.u[1] = *(const U4*)(Bb + bBase1 + ku + 8);

            acc00 = __builtin_amdgcn_wmma_f32_16x16x32_bf16(false, a0.v, false, b0.v, (short)0, acc00, false, false);
            acc01 = __builtin_amdgcn_wmma_f32_16x16x32_bf16(false, a0.v, false, b1.v, (short)0, acc01, false, false);
            acc10 = __builtin_amdgcn_wmma_f32_16x16x32_bf16(false, a1.v, false, b0.v, (short)0, acc10, false, false);
            acc11 = __builtin_amdgcn_wmma_f32_16x16x32_bf16(false, a1.v, false, b1.v, (short)0, acc11, false, false);
        }
        __syncthreads();                        // tile t consumed
    }

    // Epilogue. C layout: lane col = l, VGPR r -> row r (lanes<16) / r+8 (lanes>=16).
    const int col0 = blkN + wn * 32 + l;
    const float bv0 = bias ? bias[col0]      : 0.f;
    const float bv1 = bias ? bias[col0 + 16] : 0.f;
    #pragma unroll
    for (int mt = 0; mt < 2; ++mt) {
        const v8f* am[2] = { mt ? &acc10 : &acc00, mt ? &acc11 : &acc01 };
        const int rowb = blkM + wm * 32 + mt * 16 + hi * 8;
        #pragma unroll
        for (int r = 0; r < 8; ++r) {
            float v0 = (*am[0])[r] * scale + bv0;
            float v1 = (*am[1])[r] * scale + bv1;
            if (relu) { v0 = fmaxf(v0, 0.f); v1 = fmaxf(v1, 0.f); }
            const size_t o0 = (size_t)(rowb + r) * ldc + col0;
            if (Cf) { Cf[o0] = v0; Cf[o0 + 16] = v1; }
            if (Cb) { Cb[o0] = f2b(v0); Cb[o0 + 16] = f2b(v1); }
        }
    }
}

// ---------------------------------------------------------------------------
// f32 -> bf16 bulk conversion
// ---------------------------------------------------------------------------
__global__ void k_f2b(const float* __restrict__ in, unsigned short* __restrict__ out, int n) {
    int i = blockIdx.x * 256 + threadIdx.x;
    if (i < n) out[i] = f2b(in[i]);
}

// ---------------------------------------------------------------------------
// Embedding gather * sqrt(D) + sinusoidal PE. grid=SEQ, block=256.
// ---------------------------------------------------------------------------
__global__ void k_embed(const int* __restrict__ tokens, const float* __restrict__ emb,
                        float* __restrict__ xf, unsigned short* __restrict__ xb) {
    const int s = blockIdx.x;
    const int tok = tokens[s];
    for (int j = threadIdx.x; j < DM; j += 256) {
        const int i = j >> 1;
        const float div = __expf(-(float)(2 * i) * 0.017988946039015984f); // ln(10000)/512
        const float ang = (float)s * div;
        const float pe = (j & 1) ? __cosf(ang) : __sinf(ang);
        const float v = emb[(size_t)tok * DM + j] * 22.62741699796952f + pe; // sqrt(512)
        xf[(size_t)s * DM + j] = v;
        xb[(size_t)s * DM + j] = f2b(v);
    }
}

// ---------------------------------------------------------------------------
// Chunk-stage attention: attend over NC=4 chunks per intra-chunk position.
// grid=(CH, NH), block=HD.
// ---------------------------------------------------------------------------
__global__ void k_chunk_attn(const float* __restrict__ qkv, unsigned short* __restrict__ out) {
    __shared__ float q[NC][HD], k[NC][HD], v[NC][HD];
    __shared__ float sc[NC][NC], at[NC][NC];
    const int pos = blockIdx.x, h = blockIdx.y, t = threadIdx.x;

    for (int c = 0; c < NC; ++c) {
        const size_t r = (size_t)(c * CH + pos) * D3 + h * HD + t;
        q[c][t] = qkv[r];
        k[c][t] = qkv[r + DM];
        v[c][t] = qkv[r + 2 * DM];
    }
    __syncthreads();

    if (t < 16) {
        const int i = t >> 2, j = t & 3;
        float s = 0.f;
        for (int d = 0; d < HD; ++d) s += q[i][d] * k[j][d];
        sc[i][j] = s * 0.125f;                    // 1/sqrt(64)
    }
    __syncthreads();
    if (t < NC) {
        float m = fmaxf(fmaxf(sc[t][0], sc[t][1]), fmaxf(sc[t][2], sc[t][3]));
        float e[NC], sum = 0.f;
        for (int j = 0; j < NC; ++j) { e[j] = __expf(sc[t][j] - m); sum += e[j]; }
        const float inv = 1.f / sum;
        for (int j = 0; j < NC; ++j) at[t][j] = e[j] * inv;
    }
    __syncthreads();
    for (int i = 0; i < NC; ++i) {
        float o = 0.f;
        for (int j = 0; j < NC; ++j) o += at[i][j] * v[j][t];
        out[(size_t)(i * CH + pos) * DM + h * HD + t] = f2b(o);
    }
}

// ---------------------------------------------------------------------------
// Row softmax over n f32 scores -> normalized bf16 probs. grid=rows, block=256.
// ---------------------------------------------------------------------------
__global__ void k_softmax_row(const float* __restrict__ S, unsigned short* __restrict__ P, int n) {
    __shared__ float red[256];
    const int row = blockIdx.x, t = threadIdx.x;
    const float* s = S + (size_t)row * n;

    float m = -1e30f;
    for (int j = t; j < n; j += 256) m = fmaxf(m, s[j]);
    red[t] = m; __syncthreads();
    for (int o = 128; o; o >>= 1) { if (t < o) red[t] = fmaxf(red[t], red[t + o]); __syncthreads(); }
    m = red[0]; __syncthreads();

    float sum = 0.f;
    for (int j = t; j < n; j += 256) sum += __expf(s[j] - m);
    red[t] = sum; __syncthreads();
    for (int o = 128; o; o >>= 1) { if (t < o) red[t] += red[t + o]; __syncthreads(); }
    const float inv = 1.f / red[0];

    unsigned short* p = P + (size_t)row * n;
    for (int j = t; j < n; j += 256) p[j] = f2b(__expf(s[j] - m) * inv);
}

// ---------------------------------------------------------------------------
// Per-head V transpose: Vt[hd][key] = qkv_g[key][2D + h*HD + hd] (bf16 copy)
// ---------------------------------------------------------------------------
__global__ void k_transpose_v(const unsigned short* __restrict__ qkv_g,
                              unsigned short* __restrict__ Vt, int h) {
    const int idx = blockIdx.x * 256 + threadIdx.x;   // SEQ*HD total
    const int key = idx >> 6, hd = idx & 63;
    Vt[(size_t)hd * SEQ + key] = qkv_g[(size_t)key * D3 + 2 * DM + h * HD + hd];
}

// ---------------------------------------------------------------------------
// Fused residual + LayerNorm: y = LN(a+b)*g + beta. grid=SEQ, block=256.
// ---------------------------------------------------------------------------
__global__ void k_add_ln(const float* __restrict__ A, const float* __restrict__ B,
                         const float* __restrict__ G, const float* __restrict__ Bb,
                         float* __restrict__ Yf, unsigned short* __restrict__ Yb) {
    __shared__ float red[256];
    __shared__ float s_mean, s_rstd;
    const int row = blockIdx.x, t = threadIdx.x;
    const size_t base = (size_t)row * DM;

    const float x0 = A[base + t]       + B[base + t];
    const float x1 = A[base + t + 256] + B[base + t + 256];
    red[t] = x0 + x1; __syncthreads();
    for (int o = 128; o; o >>= 1) { if (t < o) red[t] += red[t + o]; __syncthreads(); }
    if (t == 0) s_mean = red[0] * (1.f / DM);
    __syncthreads();
    const float m = s_mean;
    const float d0 = x0 - m, d1 = x1 - m;
    red[t] = d0 * d0 + d1 * d1; __syncthreads();
    for (int o = 128; o; o >>= 1) { if (t < o) red[t] += red[t + o]; __syncthreads(); }
    if (t == 0) s_rstd = rsqrtf(red[0] * (1.f / DM) + 1e-5f);
    __syncthreads();
    const float r = s_rstd;
    const float y0 = d0 * r * G[t]       + Bb[t];
    const float y1 = d1 * r * G[t + 256] + Bb[t + 256];
    if (Yf) { Yf[base + t] = y0; Yf[base + t + 256] = y1; }
    if (Yb) { Yb[base + t] = f2b(y0); Yb[base + t + 256] = f2b(y1); }
}

// ---------------------------------------------------------------------------
// Host-side orchestration
// ---------------------------------------------------------------------------
extern "C" void kernel_launch(void* const* d_in, const int* in_sizes, int n_in,
                              void* d_out, int out_size, void* d_ws, size_t ws_size,
                              hipStream_t stream) {
    (void)in_sizes; (void)n_in; (void)out_size; (void)ws_size;

    const int*   tokens = (const int*)  d_in[0];
    const float* emb    = (const float*)d_in[1];
    const float* cw_in  = (const float*)d_in[2];
    const float* cb_in  = (const float*)d_in[3];
    const float* cw_out = (const float*)d_in[4];
    const float* cb_out = (const float*)d_in[5];
    const float* gw_in  = (const float*)d_in[6];
    const float* gb_in  = (const float*)d_in[7];
    const float* gw_out = (const float*)d_in[8];
    const float* gb_out = (const float*)d_in[9];
    const float* ln1_g  = (const float*)d_in[10];
    const float* ln1_b  = (const float*)d_in[11];
    const float* fw1    = (const float*)d_in[12];
    const float* fb1    = (const float*)d_in[13];
    const float* fw2    = (const float*)d_in[14];
    const float* fb2    = (const float*)d_in[15];
    const float* ln2_g  = (const float*)d_in[16];
    const float* ln2_b  = (const float*)d_in[17];
    const float* fcw    = (const float*)d_in[18];
    const float* fcb    = (const float*)d_in[19];
    float* logits = (float*)d_out;

    // --- workspace carve (256B aligned) ---
    char* p = (char*)d_ws;
    auto carve = [&](size_t bytes) -> void* {
        void* r = (void*)p; p += (bytes + 255) & ~(size_t)255; return r;
    };
    unsigned short* xb      = (unsigned short*)carve((size_t)SEQ * DM * 2);
    float*          xf      = (float*)         carve((size_t)SEQ * DM * 4);
    float*          qkvc    = (float*)         carve((size_t)SEQ * D3 * 4);
    unsigned short* capre_b = (unsigned short*)carve((size_t)SEQ * DM * 2);
    unsigned short* ca_b    = (unsigned short*)carve((size_t)SEQ * DM * 2);
    unsigned short* qkvg_b  = (unsigned short*)carve((size_t)SEQ * D3 * 2);
    float*          scores  = (float*)         carve((size_t)SEQ * SEQ * 4);
    unsigned short* prob_b  = (unsigned short*)carve((size_t)SEQ * SEQ * 2);
    unsigned short* vt_b    = (unsigned short*)carve((size_t)HD * SEQ * 2);
    unsigned short* gap_b   = (unsigned short*)carve((size_t)SEQ * DM * 2);
    float*          ga_f    = (float*)         carve((size_t)SEQ * DM * 4);
    float*          x1_f    = (float*)         carve((size_t)SEQ * DM * 4);
    unsigned short* x1_b    = (unsigned short*)carve((size_t)SEQ * DM * 2);
    unsigned short* ffh_b   = (unsigned short*)carve((size_t)SEQ * DF * 2);
    float*          f_f     = (float*)         carve((size_t)SEQ * DM * 4);
    unsigned short* x2_b    = (unsigned short*)carve((size_t)SEQ * DM * 2);
    unsigned short* cwin_b  = (unsigned short*)carve((size_t)D3 * DM * 2);
    unsigned short* cwout_b = (unsigned short*)carve((size_t)DM * DM * 2);
    unsigned short* gwin_b  = (unsigned short*)carve((size_t)D3 * DM * 2);
    unsigned short* gwout_b = (unsigned short*)carve((size_t)DM * DM * 2);
    unsigned short* fw1_b   = (unsigned short*)carve((size_t)DF * DM * 2);
    unsigned short* fw2_b   = (unsigned short*)carve((size_t)DM * DF * 2);
    unsigned short* fcw_b   = (unsigned short*)carve((size_t)VOC * DM * 2);

    // --- weights f32 -> bf16 ---
    auto cvt = [&](const float* src, unsigned short* dst, int n) {
        k_f2b<<<(n + 255) / 256, 256, 0, stream>>>(src, dst, n);
    };
    cvt(cw_in,  cwin_b,  D3 * DM);
    cvt(cw_out, cwout_b, DM * DM);
    cvt(gw_in,  gwin_b,  D3 * DM);
    cvt(gw_out, gwout_b, DM * DM);
    cvt(fw1,    fw1_b,   DF * DM);
    cvt(fw2,    fw2_b,   DM * DF);
    cvt(fcw,    fcw_b,   VOC * DM);

    // --- embedding + PE ---
    k_embed<<<SEQ, 256, 0, stream>>>(tokens, emb, xf, xb);

    // --- chunk QKV: (SEQ x D3) f32 ---
    k_gemm_bf16<<<dim3(D3 / 64, SEQ / 128), 256, 0, stream>>>(
        xb, DM, cwin_b, DM, cb_in, qkvc, nullptr, D3, DM, 1.f, 0);

    // --- chunk attention (L=4 over chunks) -> bf16 ---
    k_chunk_attn<<<dim3(CH, NH), HD, 0, stream>>>(qkvc, capre_b);

    // --- chunk out-proj -> ca bf16 ---
    k_gemm_bf16<<<dim3(DM / 64, SEQ / 128), 256, 0, stream>>>(
        capre_b, DM, cwout_b, DM, cb_out, nullptr, ca_b, DM, DM, 1.f, 0);

    // --- global QKV -> bf16 ---
    k_gemm_bf16<<<dim3(D3 / 64, SEQ / 128), 256, 0, stream>>>(
        ca_b, DM, gwin_b, DM, gb_in, nullptr, qkvg_b, D3, DM, 1.f, 0);

    // --- global attention, per head ---
    for (int h = 0; h < NH; ++h) {
        // scores = Q Kt / sqrt(hd): contraction over hd=64
        k_gemm_bf16<<<dim3(SEQ / 64, SEQ / 128), 256, 0, stream>>>(
            qkvg_b + h * HD, D3, qkvg_b + DM + h * HD, D3, nullptr,
            scores, nullptr, SEQ, HD, 0.125f, 0);
        k_softmax_row<<<SEQ, 256, 0, stream>>>(scores, prob_b, SEQ);
        k_transpose_v<<<(SEQ * HD) / 256, 256, 0, stream>>>(qkvg_b, vt_b, h);
        // O_head = P @ V : contraction over keys=4096, into head columns of gap_b
        k_gemm_bf16<<<dim3(HD / 64, SEQ / 128), 256, 0, stream>>>(
            prob_b, SEQ, vt_b, SEQ, nullptr, nullptr, gap_b + h * HD, DM, SEQ, 1.f, 0);
    }

    // --- global out-proj -> ga f32 ---
    k_gemm_bf16<<<dim3(DM / 64, SEQ / 128), 256, 0, stream>>>(
        gap_b, DM, gwout_b, DM, gb_out, ga_f, nullptr, DM, DM, 1.f, 0);

    // --- LN1(x + ga) ---
    k_add_ln<<<SEQ, 256, 0, stream>>>(xf, ga_f, ln1_g, ln1_b, x1_f, x1_b);

    // --- FFN ---
    k_gemm_bf16<<<dim3(DF / 64, SEQ / 128), 256, 0, stream>>>(
        x1_b, DM, fw1_b, DM, fb1, nullptr, ffh_b, DF, DM, 1.f, 1 /*relu*/);
    k_gemm_bf16<<<dim3(DM / 64, SEQ / 128), 256, 0, stream>>>(
        ffh_b, DF, fw2_b, DF, fb2, f_f, nullptr, DM, DF, 1.f, 0);

    // --- LN2(x1 + f) ---
    k_add_ln<<<SEQ, 256, 0, stream>>>(x1_f, f_f, ln2_g, ln2_b, nullptr, x2_b);

    // --- final vocab projection (dominant GEMM) ---
    k_gemm_bf16<<<dim3(VOC / 64, SEQ / 128), 256, 0, stream>>>(
        x2_b, DM, fcw_b, DM, fcb, logits, nullptr, VOC, DM, 1.f, 0);
}
